// OPTAttention_8504035246565
// MI455X (gfx1250) — matrix operand order repaired and checked by reading, not compiled, not run on hardware
//
#include <hip/hip_runtime.h>
#include <hip/hip_bf16.h>

// ---------------------------------------------------------------------------
// OPT attention (B=4,T=1024,E=2048,H=32,D=64) with 8-bit fake-quant, CDNA5.
// Matmuls: V_WMMA_F32_16X16X32_F16. GEMM tile staging: TENSOR_LOAD_TO_LDS
// (TDM) double-buffered with s_wait_tensorcnt, with LDS row padding done by
// the TDM pad feature (pad_interval=32 DW, pad_amount=4 DW -> 72-half rows).
// ---------------------------------------------------------------------------

typedef __attribute__((ext_vector_type(16))) _Float16 v16h;
typedef __attribute__((ext_vector_type(8)))  float    v8f;

#define BC 4
#define TC 1024
#define EC 2048
#define HC 32
#define DC 64

#ifndef __has_builtin
#define __has_builtin(x) 0
#endif
#if __has_builtin(__builtin_amdgcn_tensor_load_to_lds)
#define USE_TDM 1
#else
#define USE_TDM 0
#endif

union FragH { v16h v; unsigned u[8]; };

// A-matrix 16x32 f16 fragment (ISA 7.12.2): lane row = l%16,
// VGPR v holds K = (v&3)*2 + (l/16)*8 + (v>>2)*16 (+ kc*32 chunk select).
__device__ __forceinline__ v16h load_frag_a(const _Float16* base, int stride,
                                            int lane, int kc) {
  FragH f;
  const int half = lane >> 4;
  const int row  = lane & 15;
#pragma unroll
  for (int v = 0; v < 8; ++v) {
    const int k0 = ((v & 3) << 1) + (half << 3) + ((v >> 2) << 4) + (kc << 5);
    f.u[v] = *(const unsigned*)(base + row * stride + k0);
  }
  return f.v;
}

// B-matrix 32x16 f16 fragment: lane col n = l%16, VGPR v holds
// K = (v&3)*2 + (l/16)*8 + (v>>2)*16 (+ kc*32). LDS stored transposed: [n][k].
__device__ __forceinline__ v16h load_frag_b(const _Float16* base, int stride,
                                            int lane, int kc) {
  FragH f;
  const int half = lane >> 4;
  const int col  = lane & 15;
#pragma unroll
  for (int v = 0; v < 8; ++v) {
    const int k0 = ((v & 3) << 1) + (half << 3) + ((v >> 2) << 4) + (kc << 5);
    f.u[v] = *(const unsigned*)(base + col * stride + k0);
  }
  return f.v;
}

__device__ __forceinline__ unsigned lds_byte_addr(const void* p) {
  // Generic pointers to LDS carry the LDS byte offset in the low 32 bits
  // (ISA 10.2).
  return (unsigned)(unsigned long long)(uintptr_t)p;
}

#if USE_TDM
typedef unsigned int v4u __attribute__((ext_vector_type(4)));
typedef int          v8i __attribute__((ext_vector_type(8)));
typedef int          v4i __attribute__((ext_vector_type(4)));

// Issue one TDM 2D tile load: 128 rows x 64 halves from a row-major f16
// matrix with row stride gdim0 (elements), into LDS at lds_byte, padding
// each 128B row with 16B (-> 72-half LDS row stride).
__device__ __forceinline__ void tdm_load_tile_f16(const _Float16* gsrc,
                                                  unsigned lds_byte,
                                                  int gdim0, int gdim1) {
  const unsigned long long ga = (unsigned long long)(uintptr_t)gsrc;
  v4u g0;
  g0.x = 1u;                                              // count=1, user mode
  g0.y = lds_byte;                                        // lds_addr
  g0.z = (unsigned)(ga & 0xffffffffu);                    // global_addr[31:0]
  g0.w = (unsigned)((ga >> 32) & 0x01ffffffu) | (2u << 30);  // [56:32] | type=2
  const unsigned w0 = (1u << 16)      // data_size = 2 bytes
                    | (1u << 20)      // pad_enable
                    | (4u << 22)      // pad_interval: 32 DWORDs (=128B row)
                    | (3u << 25);     // pad_amount:    4 DWORDs (=16B)
  const unsigned d0 = (unsigned)gdim0, d1 = (unsigned)gdim1;
  const unsigned w1 = (d0 & 0xffffu) << 16;                    // dim0[15:0]
  const unsigned w2 = ((d0 >> 16) & 0xffffu) | ((d1 & 0xffffu) << 16);
  const unsigned w3 = ((d1 >> 16) & 0xffffu) | (64u << 16);    // tile_dim0=64
  const unsigned w4 = 128u;                                    // tile_dim1=128
  const unsigned w5 = d0;                                      // stride0 = K
  v8i g1 = {(int)w0, (int)w1, (int)w2, (int)w3, (int)w4, (int)w5, 0, 0};
  v4i z4 = {0, 0, 0, 0};
  v8i z8 = {0, 0, 0, 0, 0, 0, 0, 0};
  // 6-arg form (amdgpu-toolchain / clang-23): groups 2,3 + extra group, cpol.
  __builtin_amdgcn_tensor_load_to_lds(g0, g1, z4, z4, z8, 0);
}
#endif

// ---------------------------------------------------------------------------
// f32 -> f16 conversion (hidden + weights), vectorized.
// ---------------------------------------------------------------------------
__global__ __launch_bounds__(256)
void cvt_f32f16_kernel(const float* __restrict__ src, _Float16* __restrict__ dst,
                       long n) {
  const long i = ((long)blockIdx.x * blockDim.x + threadIdx.x) * 4;
  if (i >= n) return;
  const float4 x = *(const float4*)(src + i);
  union { _Float16 h[4]; unsigned u[2]; } p;
  p.h[0] = (_Float16)x.x; p.h[1] = (_Float16)x.y;
  p.h[2] = (_Float16)x.z; p.h[3] = (_Float16)x.w;
  *(unsigned*)(dst + i)     = p.u[0];
  *(unsigned*)(dst + i + 2) = p.u[1];
}

// ---------------------------------------------------------------------------
// GEMM: out = (A[M,K] @ W[N,K]^T + bias[N]) * scale, A/W in f16.
// Block tile 128x128, BK=64, 8 waves (2Mx4N), wave tile 64x32.
// TDM double-buffered staging (wave 0 drives the DMA engine).
// ---------------------------------------------------------------------------
__global__ __launch_bounds__(256)
void gemm_h_xwT_kernel(const _Float16* __restrict__ A,
                       const _Float16* __restrict__ W,
                       const float* __restrict__ bias, float* __restrict__ out,
                       int M, int N, int K, float scale, int permute) {
  __shared__ _Float16 As[2][128 * 72];   // [m][k], TDM-padded 64->72 halves
  __shared__ _Float16 Bs[2][128 * 72];   // [n][k]

  const int tid  = threadIdx.x;
  const int lane = tid & 31;
  const int wave = tid >> 5;
  const int wm   = wave >> 2;
  const int wn   = wave & 3;
  const int m0   = blockIdx.y * 128;
  const int n0   = blockIdx.x * 128;
  const int NK   = K >> 6;

  v8f acc[4][2];
#pragma unroll
  for (int mt = 0; mt < 4; ++mt)
#pragma unroll
    for (int nt = 0; nt < 2; ++nt)
      acc[mt][nt] = v8f{0.f, 0.f, 0.f, 0.f, 0.f, 0.f, 0.f, 0.f};

#if USE_TDM
  if (wave == 0) {
    tdm_load_tile_f16(A + (size_t)m0 * K, lds_byte_addr(&As[0][0]), K, M);
    tdm_load_tile_f16(W + (size_t)n0 * K, lds_byte_addr(&Bs[0][0]), K, N);
    if (NK > 1) {
      tdm_load_tile_f16(A + (size_t)m0 * K + 64, lds_byte_addr(&As[1][0]), K, M);
      tdm_load_tile_f16(W + (size_t)n0 * K + 64, lds_byte_addr(&Bs[1][0]), K, N);
    }
  }
#endif

  for (int it = 0; it < NK; ++it) {
    const int buf = it & 1;
#if USE_TDM
    if (wave == 0) {
      if (it + 1 < NK) __builtin_amdgcn_s_wait_tensorcnt(2);  // tile `it` done
      else             __builtin_amdgcn_s_wait_tensorcnt(0);
    }
    __syncthreads();
#else
    __syncthreads();
    // Fallback staging: cooperative f16 copy (no TDM builtin available).
#pragma unroll
    for (int i = 0; i < 16; ++i) {
      const int idx = i * 256 + tid;
      const int r = idx >> 5, c2 = (idx & 31) << 1;
      *(unsigned*)(&As[buf][r * 72 + c2]) =
          *(const unsigned*)(A + (size_t)(m0 + r) * K + it * 64 + c2);
      *(unsigned*)(&Bs[buf][r * 72 + c2]) =
          *(const unsigned*)(W + (size_t)(n0 + r) * K + it * 64 + c2);
    }
    __syncthreads();
#endif
    const _Float16* as = &As[buf][0];
    const _Float16* bs = &Bs[buf][0];
#pragma unroll
    for (int kc = 0; kc < 2; ++kc) {
      v16h af[4];
#pragma unroll
      for (int mt = 0; mt < 4; ++mt)
        af[mt] = load_frag_a(as + (wm * 64 + mt * 16) * 72, 72, lane, kc);
#pragma unroll
      for (int nt = 0; nt < 2; ++nt) {
        v16h bf = load_frag_b(bs + (wn * 32 + nt * 16) * 72, 72, lane, kc);
#pragma unroll
        for (int mt = 0; mt < 4; ++mt)
          acc[mt][nt] = __builtin_amdgcn_wmma_f32_16x16x32_f16(
              false, af[mt], false, bf, (short)0, acc[mt][nt], false, false);
      }
    }
#if USE_TDM
    __syncthreads();   // all waves done with buf before TDM overwrites it
    if (wave == 0 && it + 2 < NK) {
      tdm_load_tile_f16(A + (size_t)m0 * K + (size_t)(it + 2) * 64,
                        lds_byte_addr(&As[buf][0]), K, M);
      tdm_load_tile_f16(W + (size_t)n0 * K + (size_t)(it + 2) * 64,
                        lds_byte_addr(&Bs[buf][0]), K, N);
    }
#endif
  }

  // Epilogue: bias + scale (+ [B*T,H*D] -> [B,H,T,D] permute).
  const int half = lane >> 4;
  const int col  = lane & 15;
#pragma unroll
  for (int nt = 0; nt < 2; ++nt) {
    const int n = n0 + wn * 32 + nt * 16 + col;
    const float bn = bias[n];
#pragma unroll
    for (int mt = 0; mt < 4; ++mt) {
#pragma unroll
      for (int v = 0; v < 8; ++v) {
        const int m = m0 + wm * 64 + mt * 16 + v + (half << 3);
        const float val = (acc[mt][nt][v] + bn) * scale;
        size_t dst;
        if (permute) {
          const int b = m / TC, t = m % TC;
          const int h = n >> 6, dd = n & 63;
          dst = (((size_t)(b * HC + h) * TC) + t) * DC + dd;
        } else {
          dst = (size_t)m * N + n;
        }
        out[dst] = val;
      }
    }
  }
}

// ---------------------------------------------------------------------------
// Per-(b,h,d) stats over T: K abs-max (SmoothK), V min/max (axis=2 quant).
// ---------------------------------------------------------------------------
__global__ void colstats_kernel(const float* __restrict__ k,
                                const float* __restrict__ v,
                                float* __restrict__ sK,
                                float* __restrict__ vmn,
                                float* __restrict__ vmx, int ncols) {
  const int c = blockIdx.x * blockDim.x + threadIdx.x;
  if (c >= ncols) return;
  const int bh = c / DC, d = c % DC;
  const size_t base = (size_t)bh * TC * DC + d;
  float am = 0.f, mx = 0.f, mn = 0.f;
  for (int t = 0; t < TC; ++t) {
    const float kv = k[base + (size_t)t * DC];
    const float vv = v[base + (size_t)t * DC];
    am = fmaxf(am, fabsf(kv));
    mx = fmaxf(mx, vv);
    mn = fminf(mn, vv);
  }
  sK[c]  = fmaxf(am, 1e-5f);
  vmx[c] = fmaxf(mx, 0.f);
  vmn[c] = fminf(mn, 0.f);
}

// ---------------------------------------------------------------------------
// Row fake-quant (axis=-1 over D=64), one wave per row, 2 elems/lane.
// smooth=1: K path (divide by sK, quantize, rescale back). Output f16.
// ---------------------------------------------------------------------------
__global__ __launch_bounds__(256)
void quant_rows_kernel(const float* __restrict__ x, const float* __restrict__ sK,
                       _Float16* __restrict__ out, int smooth, long nrows) {
  const int wave = threadIdx.x >> 5;
  const int lane = threadIdx.x & 31;
  const long row = (long)blockIdx.x * 8 + wave;
  if (row >= nrows) return;
  const int d0 = lane << 1;
  float s0 = 1.f, s1 = 1.f;
  if (smooth) {
    const long bh = row / TC;
    s0 = sK[bh * DC + d0];
    s1 = sK[bh * DC + d0 + 1];
  }
  const float x0 = x[row * DC + d0] / s0;
  const float x1 = x[row * DC + d0 + 1] / s1;
  float mx = fmaxf(fmaxf(x0, x1), 0.f);
  float mn = fminf(fminf(x0, x1), 0.f);
#pragma unroll
  for (int off = 16; off >= 1; off >>= 1) {
    mx = fmaxf(mx, __shfl_xor(mx, off, 32));
    mn = fminf(mn, __shfl_xor(mn, off, 32));
  }
  float scale = (mx - mn) * (1.f / 255.f);
  if (scale <= 0.f) scale = 1.f;
  const float zero = rintf(-mn / scale);
  const float q0 = fminf(fmaxf(rintf(x0 / scale) + zero, 0.f), 255.f);
  const float q1 = fminf(fmaxf(rintf(x1 / scale) + zero, 0.f), 255.f);
  out[row * DC + d0]     = (_Float16)(scale * (q0 - zero) * s0);
  out[row * DC + d0 + 1] = (_Float16)(scale * (q1 - zero) * s1);
}

// ---------------------------------------------------------------------------
// Column fake-quant for V (axis=2 over T). One thread per (b,h,d).
// ---------------------------------------------------------------------------
__global__ void quant_cols_kernel(const float* __restrict__ v,
                                  const float* __restrict__ vmn,
                                  const float* __restrict__ vmx,
                                  _Float16* __restrict__ out, int ncols) {
  const int c = blockIdx.x * blockDim.x + threadIdx.x;
  if (c >= ncols) return;
  const float mn = vmn[c], mx = vmx[c];
  float scale = (mx - mn) * (1.f / 255.f);
  if (scale <= 0.f) scale = 1.f;
  const float zero = rintf(-mn / scale);
  const int bh = c / DC, d = c % DC;
  const size_t base = (size_t)bh * TC * DC + d;
  for (int t = 0; t < TC; ++t) {
    const float x = v[base + (size_t)t * DC];
    const float q = fminf(fmaxf(rintf(x / scale) + zero, 0.f), 255.f);
    out[base + (size_t)t * DC] = (_Float16)(scale * (q - zero));
  }
}

// ---------------------------------------------------------------------------
// Flash attention: block = one (b,h) x 64 q rows, 4 waves x 16 q rows.
// S^T = K*Q^T (lane owns one q column -> softmax stats reduce in-lane +
// one shfl_xor(16)). P round-trips via per-wave LDS tile for P*V fragments.
// Causal mask applied analytically (== adding the -1e9 mask input).
// Output stored as f16 (feeds the f16 O-projection GEMM).
// ---------------------------------------------------------------------------
__global__ __launch_bounds__(128)
void attn_kernel(const _Float16* __restrict__ Q, const _Float16* __restrict__ Kq,
                 const _Float16* __restrict__ V, _Float16* __restrict__ out) {
  __shared__ _Float16 Qs[64 * 72];      // [q][d]
  __shared__ _Float16 Ks[64 * 72];      // [k][d]
  __shared__ _Float16 Vts[64 * 72];     // [d][k]  (V transposed)
  __shared__ _Float16 Ps[4][16 * 72];   // per-wave P tile [q][k]

  const int qt   = blockIdx.x;
  const int bh   = blockIdx.y;
  const int b    = bh / HC;
  const int h    = bh % HC;
  const int tid  = threadIdx.x;
  const int wave = tid >> 5;
  const int lane = tid & 31;
  const int half = lane >> 4;
  const int col  = lane & 15;
  const size_t base = (size_t)bh * TC * DC;
  const int q0 = qt * 64;

#pragma unroll
  for (int i = 0; i < 16; ++i) {
    const int idx = i * 128 + tid;
    const int r = idx >> 5, c2 = (idx & 31) << 1;
    *(unsigned*)(Qs + r * 72 + c2) =
        *(const unsigned*)(Q + base + (size_t)(q0 + r) * DC + c2);
  }

  v8f oacc[4];
#pragma unroll
  for (int nt = 0; nt < 4; ++nt)
    oacc[nt] = v8f{0.f, 0.f, 0.f, 0.f, 0.f, 0.f, 0.f, 0.f};
  float mrun = -__builtin_inff();
  float lrun = 0.f;

  const int qglob = q0 + wave * 16 + col;

  for (int kt = 0; kt <= qt; ++kt) {
    const int k0g = kt * 64;
    __syncthreads();
#pragma unroll
    for (int i = 0; i < 16; ++i) {
      const int idx = i * 128 + tid;
      const int r = idx >> 5, c2 = (idx & 31) << 1;
      *(unsigned*)(Ks + r * 72 + c2) =
          *(const unsigned*)(Kq + base + (size_t)(k0g + r) * DC + c2);
    }
#pragma unroll
    for (int i = 0; i < 32; ++i) {
      const int e = i * 128 + tid;
      const int r = e >> 6, d = e & 63;
      Vts[d * 72 + r] = V[base + (size_t)(k0g + r) * DC + d];
    }
    if (kt < qt) {   // pull next K/V tiles toward the caches
      __builtin_prefetch(Kq + base + (size_t)(k0g + 64 + (tid >> 1)) * DC, 0, 1);
      __builtin_prefetch(V  + base + (size_t)(k0g + 64 + (tid >> 1)) * DC, 0, 1);
    }
    __syncthreads();

    v8f sacc[4];
#pragma unroll
    for (int mt = 0; mt < 4; ++mt)
      sacc[mt] = v8f{0.f, 0.f, 0.f, 0.f, 0.f, 0.f, 0.f, 0.f};
    v16h qf[2];
#pragma unroll
    for (int kc = 0; kc < 2; ++kc)
      qf[kc] = load_frag_b(Qs + (wave * 16) * 72, 72, lane, kc);
#pragma unroll
    for (int mt = 0; mt < 4; ++mt)
#pragma unroll
      for (int kc = 0; kc < 2; ++kc) {
        v16h kf = load_frag_a(Ks + (mt * 16) * 72, 72, lane, kc);
        sacc[mt] = __builtin_amdgcn_wmma_f32_16x16x32_f16(
            false, kf, false, qf[kc], (short)0, sacc[mt], false, false);
      }

    float tmax = -__builtin_inff();
#pragma unroll
    for (int mt = 0; mt < 4; ++mt)
#pragma unroll
      for (int v = 0; v < 8; ++v) {
        const int kk = k0g + mt * 16 + v + (half << 3);
        float sv = sacc[mt][v];
        if (kk > qglob) sv = -1e9f;
        sacc[mt][v] = sv;
        tmax = fmaxf(tmax, sv);
      }
    tmax = fmaxf(tmax, __shfl_xor(tmax, 16, 32));
    const float newm  = fmaxf(mrun, tmax);
    const float alpha = __expf(mrun - newm);
    float lsum = 0.f;
#pragma unroll
    for (int mt = 0; mt < 4; ++mt)
#pragma unroll
      for (int v = 0; v < 8; ++v) {
        const float p = __expf(sacc[mt][v] - newm);
        lsum += p;
        Ps[wave][col * 72 + mt * 16 + v + (half << 3)] = (_Float16)p;
      }
    lsum += __shfl_xor(lsum, 16, 32);
    lrun = lrun * alpha + lsum;
    mrun = newm;

#pragma unroll
    for (int v = 0; v < 8; ++v) {
      const float av = __shfl(alpha, v + (half << 3), 32);
#pragma unroll
      for (int nt = 0; nt < 4; ++nt) oacc[nt][v] *= av;
    }

    v16h pf[2];
#pragma unroll
    for (int kc = 0; kc < 2; ++kc)
      pf[kc] = load_frag_a(Ps[wave], 72, lane, kc);
#pragma unroll
    for (int nt = 0; nt < 4; ++nt)
#pragma unroll
      for (int kc = 0; kc < 2; ++kc) {
        v16h vf = load_frag_b(Vts + (nt * 16) * 72, 72, lane, kc);
        oacc[nt] = __builtin_amdgcn_wmma_f32_16x16x32_f16(
            false, pf[kc], false, vf, (short)0, oacc[nt], false, false);
      }
  }

#pragma unroll
  for (int v = 0; v < 8; ++v) {
    const float li = 1.f / __shfl(lrun, v + (half << 3), 32);
    const int qrow = q0 + wave * 16 + v + (half << 3);
#pragma unroll
    for (int nt = 0; nt < 4; ++nt) {
      const int dcol = nt * 16 + col;
      out[((size_t)b * TC + qrow) * EC + h * DC + dcol] =
          (_Float16)(oacc[nt][v] * li);
    }
  }
}

// ---------------------------------------------------------------------------
// Host-side launch.
// ---------------------------------------------------------------------------
extern "C" void kernel_launch(void* const* d_in, const int* in_sizes, int n_in,
                              void* d_out, int out_size, void* d_ws, size_t ws_size,
                              hipStream_t stream) {
  (void)in_sizes; (void)n_in; (void)out_size; (void)ws_size;
  const float* hidden = (const float*)d_in[0];
  const float* wq = (const float*)d_in[1];
  const float* bq = (const float*)d_in[2];
  const float* wk = (const float*)d_in[3];
  const float* bk = (const float*)d_in[4];
  const float* wv = (const float*)d_in[5];
  const float* bv = (const float*)d_in[6];
  const float* wo = (const float*)d_in[7];
  const float* bo = (const float*)d_in[8];
  // d_in[9] = attention_mask: causal mask reproduced analytically in-kernel.

  const size_t n  = (size_t)BC * HC * TC * DC;   // 8,388,608 (= B*T*E)
  const size_t EE = (size_t)EC * EC;             // 4,194,304
  const int ncols = BC * HC * DC;                // 8192

  char* ws = (char*)d_ws;
  size_t off = 0;
  float*    qf32 = (float*)(ws + off);    off += n * 4;
  float*    kf32 = (float*)(ws + off);    off += n * 4;
  float*    vf32 = (float*)(ws + off);    off += n * 4;
  _Float16* qh   = (_Float16*)(ws + off); off += n * 2;
  _Float16* kh   = (_Float16*)(ws + off); off += n * 2;
  _Float16* vh   = (_Float16*)(ws + off); off += n * 2;
  _Float16* hidh = (_Float16*)(ws + off); off += n * 2;
  _Float16* wqh  = (_Float16*)(ws + off); off += EE * 2;
  _Float16* wkh  = (_Float16*)(ws + off); off += EE * 2;
  _Float16* wvh  = (_Float16*)(ws + off); off += EE * 2;
  _Float16* woh  = (_Float16*)(ws + off); off += EE * 2;
  float*    sK   = (float*)(ws + off);    off += (size_t)ncols * 4;
  float*    vmn  = (float*)(ws + off);    off += (size_t)ncols * 4;
  float*    vmx  = (float*)(ws + off);    off += (size_t)ncols * 4;
  _Float16* aout = (_Float16*)qf32;  // reuse Q f32 region after quantization

  const int M = BC * TC;             // 4096
  const float scaling = 0.125f;      // D^-0.5

  // One-time f16 conversion of GEMM operands (halves DRAM/L2 traffic,
  // enables pure-byte TDM staging).
  cvt_f32f16_kernel<<<(int)(n / 1024), 256, 0, stream>>>(hidden, hidh, (long)n);
  cvt_f32f16_kernel<<<(int)(EE / 1024), 256, 0, stream>>>(wq, wqh, (long)EE);
  cvt_f32f16_kernel<<<(int)(EE / 1024), 256, 0, stream>>>(wk, wkh, (long)EE);
  cvt_f32f16_kernel<<<(int)(EE / 1024), 256, 0, stream>>>(wv, wvh, (long)EE);
  cvt_f32f16_kernel<<<(int)(EE / 1024), 256, 0, stream>>>(wo, woh, (long)EE);

  dim3 ggrid(EC / 128, M / 128);  // (16, 32)
  gemm_h_xwT_kernel<<<ggrid, 256, 0, stream>>>(hidh, wqh, bq, qf32, M, EC, EC, scaling, 1);
  gemm_h_xwT_kernel<<<ggrid, 256, 0, stream>>>(hidh, wkh, bk, kf32, M, EC, EC, 1.0f, 1);
  gemm_h_xwT_kernel<<<ggrid, 256, 0, stream>>>(hidh, wvh, bv, vf32, M, EC, EC, 1.0f, 1);

  colstats_kernel<<<ncols / 256, 256, 0, stream>>>(kf32, vf32, sK, vmn, vmx, ncols);

  const long nrows = (long)BC * HC * TC;   // 131072
  quant_rows_kernel<<<(int)(nrows / 8), 256, 0, stream>>>(qf32, nullptr, qh, 0, nrows);
  quant_rows_kernel<<<(int)(nrows / 8), 256, 0, stream>>>(kf32, sK, kh, 1, nrows);
  quant_cols_kernel<<<ncols / 256, 256, 0, stream>>>(vf32, vmn, vmx, vh, ncols);

  attn_kernel<<<dim3(TC / 64, BC * HC), 128, 0, stream>>>(qh, kh, vh, aout);

  gemm_h_xwT_kernel<<<ggrid, 256, 0, stream>>>(aout, woh, bo, (float*)d_out, M, EC, EC, 1.0f, 0);
}
